// GLMMoE_11227044512163
// MI455X (gfx1250) — compile-verified, matrix operand order checked by reading, not run on hardware
//
#include <hip/hip_runtime.h>
#include <hip/hip_bf16.h>
#include <cstdint>
#include <cstddef>

// ---------------------------------------------------------------------------
// GLM MoE layer for MI455X (gfx1250, wave32, WMMA).
// HBM-bound on fp32 weights (~590MB -> ~25us floor @ 23.3TB/s). Stream
// weights once, convert fp32->bf16 in registers, v_wmma_f32_16x16x32_bf16
// with fp32 accumulation. Double-buffered LDS + register prefetch keeps the
// next weight tile's HBM loads in flight during WMMA compute (1 barrier/step).
// ---------------------------------------------------------------------------

typedef __attribute__((ext_vector_type(8)))  __bf16 v8bf;
typedef __attribute__((ext_vector_type(16))) __bf16 v16bf;
typedef __attribute__((ext_vector_type(8)))  float  v8f;

#define T_TOK 512
#define H_DIM 2048
#define N_EXP 16
#define F_DIM 1408
// LDS K-tile row stride in bf16 elements: 32 data + 8 pad (80B rows, 16B aligned)
#define KP 40
#define ABUF (64 * KP)
#define BBUF (128 * KP)

// Load a 16x32 bf16 WMMA operand fragment from an LDS tile stored
// [16 rows][32 k] (row stride KP). Per ISA 7.12.2 (16-bit operand layout):
//   lanes 0-15 : row = lane,     K = {0..7, 16..23}
//   lanes 16-31: row = lane-16,  K = {8..15, 24..31}
// B operand uses the same pattern with "row" = output column (tile staged
// N-major, i.e. transposed).
__device__ inline v16bf load_frag(const __bf16* base, int lane) {
    const int r   = lane & 15;
    const int off = (lane >> 4) * 8;            // 8 bf16 = 16B
    const __bf16* p = base + r * KP + off;
    v8bf lo = *(const v8bf*)(p);                // ds_load_b128
    v8bf hi = *(const v8bf*)(p + 16);           // ds_load_b128
    return __builtin_shufflevector(lo, hi, 0,1,2,3,4,5,6,7,8,9,10,11,12,13,14,15);
}

__device__ inline unsigned pack_bf16x2(float lo, float hi) {
    __bf16 l = (__bf16)lo, h = (__bf16)hi;
    unsigned short ul, uh;
    __builtin_memcpy(&ul, &l, 2);
    __builtin_memcpy(&uh, &h, 2);
    return (unsigned)ul | ((unsigned)uh << 16);
}

// Register-stage one [32 k][128 n] fp32 weight tile (2 floats per kp slot).
__device__ inline void load_b_regs(const float* __restrict__ W, int k0, int ldw,
                                   int col, int g2, float* reg) {
    #pragma unroll
    for (int r = 0; r < 8; ++r) {
        const int kp = r * 2 + g2;              // 0..15 -> k = 2kp, 2kp+1
        reg[2 * r]     = W[(size_t)(k0 + 2 * kp)     * ldw + col];
        reg[2 * r + 1] = W[(size_t)(k0 + 2 * kp + 1) * ldw + col];
    }
}
// Convert + transpose into N-major LDS tile.
__device__ inline void store_b_lds(__bf16* B, int fl, int g2, const float* reg) {
    #pragma unroll
    for (int r = 0; r < 8; ++r) {
        const int kp = r * 2 + g2;
        *(unsigned*)&B[fl * KP + 2 * kp] = pack_bf16x2(reg[2 * r], reg[2 * r + 1]);
    }
}

// ---------------------------------------------------------------------------
// 1) fp32 -> bf16 convert of hidden states (x reused E*(F/128) times from L2)
// ---------------------------------------------------------------------------
__global__ void cvt_bf16_kernel(const float* __restrict__ x,
                                __bf16* __restrict__ y, int n) {
    for (int i = blockIdx.x * blockDim.x + threadIdx.x; i < n;
         i += gridDim.x * blockDim.x)
        y[i] = (__bf16)x[i];
}

// ---------------------------------------------------------------------------
// 2) Router: logits -> softmax -> top-4 -> renormalize -> combine[T, E]
// ---------------------------------------------------------------------------
__global__ void router_kernel(const float* __restrict__ x,
                              const float* __restrict__ wr,
                              float* __restrict__ combine) {
    const int t    = blockIdx.x;
    const int lane = threadIdx.x & 31;
    const int wv   = threadIdx.x >> 5;
    __shared__ float logits_s[N_EXP];

    for (int e = wv; e < N_EXP; e += 8) {
        float s = 0.f;
        const float* xr = x  + (size_t)t * H_DIM;
        const float* we = wr + (size_t)e * H_DIM;
        for (int h = lane; h < H_DIM; h += 32) s += xr[h] * we[h];
        #pragma unroll
        for (int off = 16; off > 0; off >>= 1) s += __shfl_xor(s, off, 32);
        if (lane == 0) logits_s[e] = s;
    }
    __syncthreads();

    if (threadIdx.x == 0) {
        float p[N_EXP];
        float mx = -1e30f;
        for (int e = 0; e < N_EXP; ++e) mx = fmaxf(mx, logits_s[e]);
        float sum = 0.f;
        for (int e = 0; e < N_EXP; ++e) { p[e] = __expf(logits_s[e] - mx); sum += p[e]; }
        for (int e = 0; e < N_EXP; ++e) p[e] /= sum;
        bool picked[N_EXP];
        for (int e = 0; e < N_EXP; ++e) picked[e] = false;
        float s4 = 0.f;
        for (int k = 0; k < 4; ++k) {
            int best = -1; float bv = -1.f;
            for (int e = 0; e < N_EXP; ++e)
                if (!picked[e] && p[e] > bv) { bv = p[e]; best = e; }
            picked[best] = true; s4 += bv;
        }
        for (int e = 0; e < N_EXP; ++e)
            combine[(size_t)t * N_EXP + e] = picked[e] ? p[e] / s4 : 0.f;
    }
}

// ---------------------------------------------------------------------------
// 3) Gate+Up GEMMs + SwiGLU epilogue.
//    WG tile: M=64 tokens x N=128 F-cols, K-step 32 over H.
//    8 waves as 4(M) x 2(N); each wave: 16x64 via 4 gate + 4 up accumulators.
//    Pipeline per step: store regs->LDS[buf], barrier, prefetch next tile
//    (HBM loads in flight during compute), WMMA from LDS[buf].
// ---------------------------------------------------------------------------
template <bool HAS_COMBINE>
__global__ void __launch_bounds__(256)
moe_gate_up(const __bf16* __restrict__ xbf,       // [T, H] bf16
            const float*  __restrict__ wg,        // gate weights, row stride ldw
            const float*  __restrict__ wu,        // up weights,   row stride ldw
            size_t wstride, int ldw,              // per-expert elem stride / row stride
            const float*  __restrict__ combine,   // [T, N_EXP] (unused if !HAS_COMBINE)
            __bf16* __restrict__ Aout,            // [E, T, F] bf16 (e = blockIdx.z)
            float scale) {
    __shared__ __bf16 A_lds [2 * ABUF];
    __shared__ __bf16 Bg_lds[2 * BBUF];
    __shared__ __bf16 Bu_lds[2 * BBUF];

    const int e   = blockIdx.z;
    const int m0  = blockIdx.x * 64;
    const int f0  = blockIdx.y * 128;
    const int tid = threadIdx.x;
    const int lane = tid & 31;
    const int wv   = tid >> 5;
    const int wm   = wv >> 1;   // 0..3  (16-row subtile)
    const int wn   = wv & 1;    // 0..1  (64-col group)
    // staging roles
    const int sm   = tid >> 2;  // A row 0..63
    const int sch  = tid & 3;   // A 16B chunk
    const int fl   = tid & 127; // B column
    const int g2   = tid >> 7;  // B kp phase

    const float*  wg_e   = wg + (size_t)e * wstride;
    const float*  wu_e   = wu + (size_t)e * wstride;
    __bf16*       Aout_e = Aout + (size_t)e * T_TOK * F_DIM;

    v8f accg[4] = {}, accu[4] = {};

    const int KS = H_DIM / 32;
    v8bf  a_reg;
    float bg_reg[16], bu_reg[16];

    // prologue: stage step 0 into registers
    a_reg = *(const v8bf*)&xbf[(size_t)(m0 + sm) * H_DIM + sch * 8];
    load_b_regs(wg_e, 0, ldw, f0 + fl, g2, bg_reg);
    load_b_regs(wu_e, 0, ldw, f0 + fl, g2, bu_reg);

    for (int kk = 0; kk < KS; ++kk) {
        __bf16* Ab  = A_lds  + (kk & 1) * ABUF;
        __bf16* Bgb = Bg_lds + (kk & 1) * BBUF;
        __bf16* Bub = Bu_lds + (kk & 1) * BBUF;

        // commit staged registers to this step's LDS buffer
        *(v8bf*)&Ab[sm * KP + sch * 8] = a_reg;
        store_b_lds(Bgb, fl, g2, bg_reg);
        store_b_lds(Bub, fl, g2, bu_reg);
        __syncthreads();

        // prefetch next step's tiles (HBM loads overlap the WMMAs below)
        if (kk + 1 < KS) {
            const int k1 = (kk + 1) * 32;
            a_reg = *(const v8bf*)&xbf[(size_t)(m0 + sm) * H_DIM + k1 + sch * 8];
            load_b_regs(wg_e, k1, ldw, f0 + fl, g2, bg_reg);
            load_b_regs(wu_e, k1, ldw, f0 + fl, g2, bu_reg);
        }

        // WMMA from LDS[buf]
        v16bf a = load_frag(Ab + (wm * 16) * KP, lane);
        #pragma unroll
        for (int j = 0; j < 4; ++j) {
            v16bf bg = load_frag(Bgb + (wn * 64 + j * 16) * KP, lane);
            accg[j] = __builtin_amdgcn_wmma_f32_16x16x32_bf16(
                false, a, false, bg, (short)0, accg[j], false, false);
            v16bf bu = load_frag(Bub + (wn * 64 + j * 16) * KP, lane);
            accu[j] = __builtin_amdgcn_wmma_f32_16x16x32_bf16(
                false, a, false, bu, (short)0, accu[j], false, false);
        }
    }

    // ---- epilogue: silu(g)*u * combine*scale -> bf16 ----
    const int col = lane & 15;
    const int rh  = lane >> 4;
    float cmb8[8];
    #pragma unroll
    for (int v = 0; v < 8; ++v) {
        const int t = m0 + wm * 16 + v + rh * 8;
        cmb8[v] = HAS_COMBINE ? combine[(size_t)t * N_EXP + e] * scale : scale;
    }
    #pragma unroll
    for (int j = 0; j < 4; ++j) {
        #pragma unroll
        for (int v = 0; v < 8; ++v) {
            const int t = m0 + wm * 16 + v + rh * 8;
            const int f = f0 + wn * 64 + j * 16 + col;
            const float g = accg[j][v];
            const float u = accu[j][v];
            const float sg = g / (1.f + __expf(-g));
            Aout_e[(size_t)t * F_DIM + f] = (__bf16)(sg * u * cmb8[v]);
        }
    }
}

// ---------------------------------------------------------------------------
// 4) Down projection: out[t,h] += sum_e A_e[t,:] @ Wd_e[:,h]
//    Same double-buffered pipeline; expert loop flattened into the step loop
//    (epb experts accumulate in registers), one fp32 atomicAdd per element.
// ---------------------------------------------------------------------------
__global__ void __launch_bounds__(256)
moe_down(const __bf16* __restrict__ Ain,   // [.., T, F] bf16
         const float*  __restrict__ Wd,    // [.., F, H] fp32
         float* __restrict__ out,          // [T, H] fp32, atomic accumulate
         int epb, size_t Astride, size_t Wstride) {
    __shared__ __bf16 A_lds[2 * ABUF];
    __shared__ __bf16 B_lds[2 * BBUF];

    const int m0  = blockIdx.x * 64;
    const int h0  = blockIdx.y * 128;
    const int e0  = blockIdx.z * epb;
    const int tid = threadIdx.x;
    const int lane = tid & 31;
    const int wv   = tid >> 5;
    const int wm   = wv >> 1;
    const int wn   = wv & 1;
    const int sm   = tid >> 2;
    const int sch  = tid & 3;
    const int fl   = tid & 127;
    const int g2   = tid >> 7;

    const int KS    = F_DIM / 32;
    const int total = epb * KS;

    v8f acc[4] = {};
    v8bf  a_reg;
    float b_reg[16];

    // step s -> expert e0 + s/KS, k0 = (s%KS)*32
    {
        const __bf16* Ae = Ain + (size_t)e0 * Astride;
        const float*  We = Wd  + (size_t)e0 * Wstride;
        a_reg = *(const v8bf*)&Ae[(size_t)(m0 + sm) * F_DIM + sch * 8];
        load_b_regs(We, 0, H_DIM, h0 + fl, g2, b_reg);
    }

    for (int s = 0; s < total; ++s) {
        __bf16* Ab = A_lds + (s & 1) * ABUF;
        __bf16* Bb = B_lds + (s & 1) * BBUF;

        *(v8bf*)&Ab[sm * KP + sch * 8] = a_reg;
        store_b_lds(Bb, fl, g2, b_reg);
        __syncthreads();

        if (s + 1 < total) {
            const int ei = (s + 1) / KS;
            const int k1 = ((s + 1) % KS) * 32;
            const __bf16* Ae = Ain + (size_t)(e0 + ei) * Astride;
            const float*  We = Wd  + (size_t)(e0 + ei) * Wstride;
            a_reg = *(const v8bf*)&Ae[(size_t)(m0 + sm) * F_DIM + k1 + sch * 8];
            load_b_regs(We, k1, H_DIM, h0 + fl, g2, b_reg);
        }

        v16bf a = load_frag(Ab + (wm * 16) * KP, lane);
        #pragma unroll
        for (int j = 0; j < 4; ++j) {
            v16bf b = load_frag(Bb + (wn * 64 + j * 16) * KP, lane);
            acc[j] = __builtin_amdgcn_wmma_f32_16x16x32_bf16(
                false, a, false, b, (short)0, acc[j], false, false);
        }
    }

    const int col = lane & 15;
    const int rh  = lane >> 4;
    #pragma unroll
    for (int j = 0; j < 4; ++j) {
        #pragma unroll
        for (int v = 0; v < 8; ++v) {
            const int t = m0 + wm * 16 + v + rh * 8;
            const int h = h0 + wn * 64 + j * 16 + col;
            atomicAdd(out + (size_t)t * H_DIM + h, acc[j][v]);
        }
    }
}

// ---------------------------------------------------------------------------
// Launch
// ---------------------------------------------------------------------------
extern "C" void kernel_launch(void* const* d_in, const int* in_sizes, int n_in,
                              void* d_out, int out_size, void* d_ws, size_t ws_size,
                              hipStream_t stream) {
    const float* x     = (const float*)d_in[0];  // [T, H]
    const float* wr    = (const float*)d_in[1];  // [E, H]
    const float* wg    = (const float*)d_in[2];  // [E, H, F]
    const float* wu    = (const float*)d_in[3];  // [E, H, F]
    const float* wd    = (const float*)d_in[4];  // [E, F, H]
    const float* wsin  = (const float*)d_in[5];  // [H, 2F]
    const float* wsout = (const float*)d_in[6];  // [F, H]
    float* out = (float*)d_out;
    (void)in_sizes; (void)n_in; (void)out_size;

    // Workspace layout (~26.6 MB total)
    char* ws = (char*)d_ws;
    size_t off = 0;
    auto take = [&](size_t bytes) {
        size_t o = off;
        off = (off + bytes + 255) & ~(size_t)255;
        return o;
    };
    __bf16* xbf     = (__bf16*)(ws + take((size_t)T_TOK * H_DIM * 2));
    float*  combine = (float*) (ws + take((size_t)T_TOK * N_EXP * 4));
    __bf16* Arout   = (__bf16*)(ws + take((size_t)N_EXP * T_TOK * F_DIM * 2));
    __bf16* Ashared = (__bf16*)(ws + take((size_t)T_TOK * F_DIM * 2));
    (void)ws_size;

    hipMemsetAsync(d_out, 0, (size_t)T_TOK * H_DIM * sizeof(float), stream);

    cvt_bf16_kernel<<<dim3(1024), dim3(256), 0, stream>>>(x, xbf, T_TOK * H_DIM);
    router_kernel<<<dim3(T_TOK), dim3(256), 0, stream>>>(x, wr, combine);

    // Routed experts: grid (T/64, F/128, E)
    moe_gate_up<true><<<dim3(8, 11, 16), dim3(256), 0, stream>>>(
        xbf, wg, wu, (size_t)H_DIM * F_DIM, F_DIM, combine, Arout, 2.5f);
    // Shared expert: ws_in = [H, 2F], gate = cols [0,F), up = cols [F,2F)
    moe_gate_up<false><<<dim3(8, 11, 1), dim3(256), 0, stream>>>(
        xbf, wsin, wsin + F_DIM, (size_t)0, 2 * F_DIM, nullptr, Ashared, 1.0f);

    // Routed down: 4 experts per block, grid (T/64, H/128, 4)
    moe_down<<<dim3(8, 16, 4), dim3(256), 0, stream>>>(
        Arout, wd, out, 4, (size_t)T_TOK * F_DIM, (size_t)F_DIM * H_DIM);
    // Shared down
    moe_down<<<dim3(8, 16, 1), dim3(256), 0, stream>>>(
        Ashared, wsout, out, 1, (size_t)0, (size_t)0);
}
